// BimModel_10574209483597
// MI455X (gfx1250) — matrix operand order repaired and checked by reading, not compiled, not run on hardware
//
#include <hip/hip_runtime.h>
#include <cstdint>
#include <cmath>

// ---------------------------------------------------------------------------
// MI455X (gfx1250) implementation.
// Compute-bound workload (~196 GFLOP vs ~350MB traffic @ 23.3TB/s) -> route
// all big GEMMs through v_wmma_f32_16x16x32_bf16 (wave32, 16x16x32 tiles).
// Staging uses gfx1250 async global->LDS copies (ASYNCcnt) with LDS double
// buffering so global fetch of tile k+1 overlaps WMMA on tile k.
// ---------------------------------------------------------------------------

typedef __bf16 bf16_t;
typedef __attribute__((ext_vector_type(16))) __bf16 bf16x16;
typedef __attribute__((ext_vector_type(8)))  __bf16 bf16x8;
typedef __attribute__((ext_vector_type(8)))  float  v8f;

#define BATCH   8192
#define CTX     96
#define KPAD    128        // 96 + 2 features, padded to 128 for K-tiling
#define MEMD    1536
#define MEMN    512
#define HALLN   4608       // 3 * MEMD
#define PREDL   24
#define HIDDEN  64
#define TOPK    8

__device__ __forceinline__ float softplusf(float x) {
    return (x > 20.0f) ? x : log1pf(expf(x));
}

// ---------------------------------------------------------------------------
// WMMA GEMM: C[M,N] = alpha * A[M,K](bf16,lda) @ B[K,N](bf16,ldb) + bias[N]
// Block: 256 threads (8 waves). Block tile 128x64, BK=64. Each wave: 32x32.
// Double-buffered LDS; tiles staged with global_load_async_to_lds_b128.
// ---------------------------------------------------------------------------
template <int OUT_BF16>
__global__ __launch_bounds__(256)
void gemm_bf16_wmma(const bf16_t* __restrict__ A, int lda,
                    const bf16_t* __restrict__ B, int ldb,
                    void* __restrict__ Cv, int ldc,
                    const float* __restrict__ bias,
                    float alpha, int K)
{
    constexpr int BM = 128, BN = 64, BK = 64;
    constexpr int LA = 80;   // LDS row stride (elems): 160B rows -> 16B aligned, bank-spread
    constexpr int LB = 80;

    __shared__ bf16_t sA[2][BM * LA];
    __shared__ bf16_t sB[2][BK * LB];

    const int tid  = threadIdx.x;
    const int lane = tid & 31;
    const int wave = tid >> 5;        // 0..7
    const int wm   = wave & 3;        // wave M tile (4 along M)
    const int wn   = wave >> 2;       // wave N tile (2 along N)
    const long bm  = (long)blockIdx.x * BM;
    const long bn  = (long)blockIdx.y * BN;

    v8f acc[2][2];
    #pragma unroll
    for (int mt = 0; mt < 2; ++mt)
        #pragma unroll
        for (int nt = 0; nt < 2; ++nt)
            #pragma unroll
            for (int e = 0; e < 8; ++e) acc[mt][nt][e] = 0.0f;

    // Async-stage one BK tile (A: 4 x 16B per thread, B: 2 x 16B per thread)
    // directly into LDS via the CDNA5 async data path (no VGPR round trip).
    auto stage_async = [&](int buf, int k0) {
        #pragma unroll
        for (int i = 0; i < 4; ++i) {
            int c   = tid + i * 256;       // 0..1023 chunks (8 per 64-elem row)
            int row = c >> 3;
            int col = (c & 7) << 3;
            unsigned long long g =
                (unsigned long long)(const void*)(A + (bm + row) * (long)lda + k0 + col);
            unsigned lo = (unsigned)(size_t)&sA[buf][row * LA + col];
            asm volatile("global_load_async_to_lds_b128 %0, %1, off"
                         :: "v"(lo), "v"(g) : "memory");
        }
        #pragma unroll
        for (int i = 0; i < 2; ++i) {
            int c   = tid + i * 256;       // 0..511 chunks
            int row = c >> 3;
            int col = (c & 7) << 3;
            unsigned long long g =
                (unsigned long long)(const void*)(B + (long)(k0 + row) * ldb + bn + col);
            unsigned lo = (unsigned)(size_t)&sB[buf][row * LB + col];
            asm volatile("global_load_async_to_lds_b128 %0, %1, off"
                         :: "v"(lo), "v"(g) : "memory");
        }
    };

    // Prologue: stage tile 0, wait, barrier.
    stage_async(0, 0);
    asm volatile("s_wait_asynccnt 0x0" ::: "memory");
    __syncthreads();

    int buf = 0;
    for (int k0 = 0; k0 < K; k0 += BK) {
        const bool more = (k0 + BK) < K;
        // Kick off async fetch of the next tile into the other buffer.
        if (more) stage_async(buf ^ 1, k0 + BK);

        // Compute on current buffer: 2 K-steps x 4 WMMA.
        #pragma unroll
        for (int ks = 0; ks < 2; ++ks) {
            const int r  = lane & 15;
            const int kb = (lane >> 4) << 3;   // 0 or 8 (K-half per lane group)
            bf16x16 afrag[2];
            bf16x16 bfrag[2];
            // A fragment 16x32: lane row r, elems 0..7 -> K=kb..kb+7, 8..15 -> K=kb+16..kb+23
            #pragma unroll
            for (int mt = 0; mt < 2; ++mt) {
                const bf16_t* p = &sA[buf][(wm * 32 + mt * 16 + r) * LA + ks * 32 + kb];
                bf16x8 plo = *(const bf16x8*)(p);
                bf16x8 phi = *(const bf16x8*)(p + 16);
                #pragma unroll
                for (int e = 0; e < 8; ++e) { afrag[mt][e] = plo[e]; afrag[mt][8 + e] = phi[e]; }
            }
            // B fragment 32x16: lane col r, elems 0..7 -> K=kb..kb+7, 8..15 -> K=kb+16..kb+23
            #pragma unroll
            for (int nt = 0; nt < 2; ++nt) {
                const bf16_t* q = &sB[buf][(ks * 32 + kb) * LB + wn * 32 + nt * 16 + r];
                #pragma unroll
                for (int e = 0; e < 8; ++e) { bfrag[nt][e] = q[e * LB]; bfrag[nt][8 + e] = q[(16 + e) * LB]; }
            }
            #pragma unroll
            for (int mt = 0; mt < 2; ++mt)
                #pragma unroll
                for (int nt = 0; nt < 2; ++nt)
                    acc[mt][nt] = __builtin_amdgcn_wmma_f32_16x16x32_bf16(
                        false, afrag[mt], false, bfrag[nt],
                        (short)0, acc[mt][nt], false, false);
        }

        // Next buffer must be resident in LDS before anyone reads it.
        if (more) asm volatile("s_wait_asynccnt 0x0" ::: "memory");
        __syncthreads();
        buf ^= 1;
    }

    // Epilogue. C frag layout: VGPR e -> M=e (lanes 0-15) / M=e+8 (lanes 16-31), N=lane&15.
    const int cn = lane & 15;
    const int cm = (lane >> 4) << 3;
    #pragma unroll
    for (int mt = 0; mt < 2; ++mt) {
        #pragma unroll
        for (int nt = 0; nt < 2; ++nt) {
            const long ncol = bn + wn * 32 + nt * 16 + cn;
            const long mrow = bm + wm * 32 + mt * 16 + cm;
            const float bv  = bias ? bias[ncol] : 0.0f;
            #pragma unroll
            for (int e = 0; e < 8; ++e) {
                float v   = acc[mt][nt][e] * alpha + bv;
                long  row = mrow + e;
                if (OUT_BF16) ((bf16_t*)Cv)[row * (long)ldc + ncol] = (bf16_t)v;
                else          ((float*) Cv)[row * (long)ldc + ncol] = v;
            }
        }
    }
}

// ---------------------------------------------------------------------------
// Mean scaler pass 1: per-row weighted abs-sum / obs count + global totals.
// ---------------------------------------------------------------------------
__global__ void k_zero2(float* g) { if (threadIdx.x < 2) g[threadIdx.x] = 0.0f; }

__global__ void k_scaler_pass1(const float* __restrict__ x, const float* __restrict__ w,
                               float* __restrict__ rowsum, float* __restrict__ rowobs,
                               float* __restrict__ gsum)
{
    int b = blockIdx.x * blockDim.x + threadIdx.x;
    if (b >= BATCH) return;
    float ts = 0.0f, no = 0.0f;
    #pragma unroll 4
    for (int c = 0; c < CTX; ++c) {
        float wv = w[b * CTX + c];
        ts += fabsf(x[b * CTX + c]) * wv;
        no += wv;
    }
    rowsum[b] = ts;
    rowobs[b] = no;
    atomicAdd(&gsum[0], ts);   // only feeds default_scale (selected iff num_obs==0)
    atomicAdd(&gsum[1], no);
}

// Build feat[B,128] = [scaled(96), 0, log(scale), pad 0..], emit loc/scale outputs.
__global__ void k_build_feat(const float* __restrict__ x,
                             const float* __restrict__ rowsum, const float* __restrict__ rowobs,
                             const float* __restrict__ gsum,
                             bf16_t* __restrict__ feat,
                             float* __restrict__ out_loc, float* __restrict__ out_scale)
{
    int idx = blockIdx.x * blockDim.x + threadIdx.x;
    if (idx >= BATCH * KPAD) return;
    int b = idx >> 7;
    int j = idx & 127;
    float no = rowobs[b];
    float s  = (no > 0.0f) ? (rowsum[b] / fmaxf(no, 1.0f))
                           : (gsum[0] / fmaxf(gsum[1], 1.0f));
    s = fmaxf(s, 1e-10f);
    float v;
    if (j < CTX)       v = x[b * CTX + j] / s;
    else if (j == 97)  v = logf(s);
    else               v = 0.0f;   // j==96: sign(0)*log1p(0) == 0 ; j>=98: zero pad
    feat[idx] = (bf16_t)v;
    if (j == 0) { out_loc[b] = 0.0f; out_scale[b] = s; }
}

// ---------------------------------------------------------------------------
// Weight preparation kernels (f32 -> bf16, with pad / transpose / row-norm).
// ---------------------------------------------------------------------------
__global__ void k_wb_pad2(const float* __restrict__ src, bf16_t* __restrict__ dst)
{
    int idx = blockIdx.x * blockDim.x + threadIdx.x;
    if (idx >= KPAD * MEMD) return;
    int r = idx / MEMD, c = idx % MEMD;
    dst[idx] = (bf16_t)((r < CTX + 2) ? src[r * MEMD + c] : 0.0f);
}

__global__ void k_f32_to_bf16(const float* __restrict__ src, bf16_t* __restrict__ dst, long n)
{
    long idx = (long)blockIdx.x * blockDim.x + threadIdx.x;
    if (idx < n) dst[idx] = (bf16_t)src[idx];
}

// dst[S,R] = src[R,S]^T  (dst linear idx = s*R + r)
__global__ void k_transpose_bf16(const float* __restrict__ src, bf16_t* __restrict__ dst,
                                 int R, int S)
{
    long idx = (long)blockIdx.x * blockDim.x + threadIdx.x;
    if (idx >= (long)R * S) return;
    int s = (int)(idx / R);
    int r = (int)(idx % R);
    dst[idx] = (bf16_t)src[(long)r * S + s];
}

__global__ __launch_bounds__(256) void k_ep_rownorm(const float* __restrict__ ep,
                                                    float* __restrict__ epn)
{
    int j = blockIdx.x, t = threadIdx.x;
    __shared__ float red[256];
    float ss = 0.0f;
    for (int k = t; k < MEMD; k += 256) { float v = ep[(long)j * MEMD + k]; ss += v * v; }
    red[t] = ss; __syncthreads();
    for (int s = 128; s > 0; s >>= 1) { if (t < s) red[t] += red[t + s]; __syncthreads(); }
    if (t == 0) epn[j] = sqrtf(red[0]);
}

// EpT[1536,512] = (ep / max(||ep_row||, eps))^T in bf16
__global__ void k_ep_transpose_norm(const float* __restrict__ ep, const float* __restrict__ epn,
                                    bf16_t* __restrict__ dst)
{
    int idx = blockIdx.x * blockDim.x + threadIdx.x;
    if (idx >= MEMD * MEMN) return;
    int k = idx / MEMN, j = idx % MEMN;
    dst[idx] = (bf16_t)(ep[(long)j * MEMD + k] / fmaxf(epn[j], 1e-6f));
}

// ---------------------------------------------------------------------------
// Post-backbone: pack h into h_all (bf16) + per-row L2 norm for cosine sim.
// ---------------------------------------------------------------------------
__global__ __launch_bounds__(256) void k_post_h(const float* __restrict__ h,
                                                bf16_t* __restrict__ hall,
                                                float* __restrict__ rownorm)
{
    int b = blockIdx.x, t = threadIdx.x;
    __shared__ float red[256];
    float ss = 0.0f;
    for (int j = t; j < MEMD; j += 256) {
        float v = h[(long)b * MEMD + j];
        hall[(long)b * HALLN + MEMD + j] = (bf16_t)v;
        ss += v * v;
    }
    red[t] = ss; __syncthreads();
    for (int s = 128; s > 0; s >>= 1) { if (t < s) red[t] += red[t + s]; __syncthreads(); }
    if (t == 0) rownorm[b] = sqrtf(red[0]);
}

// ---------------------------------------------------------------------------
// Softmax over 512 logits per row -> bf16 attention weights.
// ---------------------------------------------------------------------------
__global__ __launch_bounds__(256) void k_softmax512(const float* __restrict__ S,
                                                    bf16_t* __restrict__ att)
{
    int b = blockIdx.x, t = threadIdx.x;
    __shared__ float red[256];
    float v0 = S[(long)b * MEMN + t];
    float v1 = S[(long)b * MEMN + 256 + t];
    red[t] = fmaxf(v0, v1); __syncthreads();
    for (int s = 128; s > 0; s >>= 1) { if (t < s) red[t] = fmaxf(red[t], red[t + s]); __syncthreads(); }
    float mx = red[0]; __syncthreads();
    float e0 = expf(v0 - mx), e1 = expf(v1 - mx);
    red[t] = e0 + e1; __syncthreads();
    for (int s = 128; s > 0; s >>= 1) { if (t < s) red[t] += red[t + s]; __syncthreads(); }
    float inv = 1.0f / red[0];
    att[(long)b * MEMN + t]       = (bf16_t)(e0 * inv);
    att[(long)b * MEMN + 256 + t] = (bf16_t)(e1 * inv);
}

// ---------------------------------------------------------------------------
// Episodic retrieval: top-8 of sim (order invariant to /||h||), softmax of
// sim/||h||, weighted gather of raw episodic rows -> bf16 into h_all[...,2].
// ---------------------------------------------------------------------------
__global__ __launch_bounds__(256) void k_topk_ep(const float* __restrict__ sim,
                                                 const float* __restrict__ rownorm,
                                                 const float* __restrict__ ep,
                                                 bf16_t* __restrict__ hall)
{
    int b = blockIdx.x, t = threadIdx.x;
    __shared__ float ssim[MEMN];
    __shared__ float sval[256];
    __shared__ int   sidx[256];
    __shared__ float topv[TOPK];
    __shared__ int   topi[TOPK];
    __shared__ float wsh[TOPK];

    ssim[t]       = sim[(long)b * MEMN + t];
    ssim[t + 256] = sim[(long)b * MEMN + 256 + t];
    __syncthreads();

    for (int sel = 0; sel < TOPK; ++sel) {
        float v0 = ssim[t], v1 = ssim[t + 256];
        float bv; int bi;
        if (v0 >= v1) { bv = v0; bi = t; } else { bv = v1; bi = t + 256; }
        sval[t] = bv; sidx[t] = bi;
        __syncthreads();
        for (int s = 128; s > 0; s >>= 1) {
            if (t < s) {
                float ov = sval[t + s]; int oi = sidx[t + s];
                if (ov > sval[t] || (ov == sval[t] && oi < sidx[t])) { sval[t] = ov; sidx[t] = oi; }
            }
            __syncthreads();
        }
        if (t == 0) { topv[sel] = sval[0]; topi[sel] = sidx[0]; ssim[sidx[0]] = -3.4e38f; }
        __syncthreads();
    }

    if (t == 0) {
        float nrm = fmaxf(rownorm[b], 1e-6f);
        float ks[TOPK], mx = -3.4e38f;
        for (int i = 0; i < TOPK; ++i) { ks[i] = topv[i] / nrm; mx = fmaxf(mx, ks[i]); }
        float sum = 0.0f;
        for (int i = 0; i < TOPK; ++i) { ks[i] = expf(ks[i] - mx); sum += ks[i]; }
        for (int i = 0; i < TOPK; ++i) wsh[i] = ks[i] / sum;
    }
    __syncthreads();

    for (int j = t; j < MEMD; j += 256) {
        float a = 0.0f;
        #pragma unroll
        for (int i = 0; i < TOPK; ++i) a += wsh[i] * ep[(long)topi[i] * MEMD + j];
        hall[(long)b * HALLN + 2 * MEMD + j] = (bf16_t)a;   // GAMMA == 1.0
    }
}

// ---------------------------------------------------------------------------
// StudentT head: y[B,1536] -> (df, loc, scale)[B,24,3]
// ---------------------------------------------------------------------------
__global__ void k_proj(const float* __restrict__ y, const float* __restrict__ Wp,
                       const float* __restrict__ bp, float* __restrict__ out)
{
    int i = blockIdx.x * blockDim.x + threadIdx.x;   // i = b*24 + p
    if (i >= BATCH * PREDL) return;
    const float* yr = y + (long)i * HIDDEN;          // b*1536 + p*64 == i*64
    float a0 = bp[0], a1 = bp[1], a2 = bp[2];
    #pragma unroll 8
    for (int k = 0; k < HIDDEN; ++k) {
        float v = yr[k];
        a0 += v * Wp[k * 3 + 0];
        a1 += v * Wp[k * 3 + 1];
        a2 += v * Wp[k * 3 + 2];
    }
    out[(long)i * 3 + 0] = 2.0f + softplusf(a0);
    out[(long)i * 3 + 1] = a1;
    out[(long)i * 3 + 2] = softplusf(a2);
}

// ---------------------------------------------------------------------------
// Host orchestration
// ---------------------------------------------------------------------------
extern "C" void kernel_launch(void* const* d_in, const int* in_sizes, int n_in,
                              void* d_out, int out_size, void* d_ws, size_t ws_size,
                              hipStream_t stream)
{
    (void)in_sizes; (void)n_in; (void)out_size; (void)ws_size;

    const float* past   = (const float*)d_in[0];
    const float* obs    = (const float*)d_in[1];
    const float* Wb     = (const float*)d_in[2];
    const float* bb     = (const float*)d_in[3];
    const float* Wq     = (const float*)d_in[4];
    const float* Mem    = (const float*)d_in[5];
    const float* Ep     = (const float*)d_in[6];
    const float* Wend   = (const float*)d_in[7];
    const float* bend   = (const float*)d_in[8];
    const float* Wproj  = (const float*)d_in[9];
    const float* bproj  = (const float*)d_in[10];
    float*       out    = (float*)d_out;

    float* out_loc   = out + (long)BATCH * PREDL * 3;          // [B]
    float* out_scale = out_loc + BATCH;                        // [B]

    // ---- workspace carve-up (256B aligned) ----
    char*  wsb = (char*)d_ws;
    size_t off = 0;
    auto carve = [&](size_t bytes) -> void* {
        off = (off + 255) & ~(size_t)255;
        void* p = wsb + off;
        off += bytes;
        return p;
    };
    float*  rowsum   = (float*) carve(BATCH * 4);
    float*  rowobs   = (float*) carve(BATCH * 4);
    float*  gsum     = (float*) carve(2 * 4);
    float*  rownorm  = (float*) carve(BATCH * 4);
    float*  epnorm   = (float*) carve(MEMN * 4);
    bf16_t* feat     = (bf16_t*)carve((size_t)BATCH * KPAD * 2);
    bf16_t* Wb_b     = (bf16_t*)carve((size_t)KPAD * MEMD * 2);
    bf16_t* Wq_b     = (bf16_t*)carve((size_t)MEMD * MEMD * 2);
    bf16_t* MemT_b   = (bf16_t*)carve((size_t)MEMD * MEMN * 2);
    bf16_t* Mem_b    = (bf16_t*)carve((size_t)MEMN * MEMD * 2);
    bf16_t* EpT_b    = (bf16_t*)carve((size_t)MEMD * MEMN * 2);
    bf16_t* WendT_b  = (bf16_t*)carve((size_t)HALLN * MEMD * 2);
    float*  hbuf     = (float*) carve((size_t)BATCH * MEMD * 4);   // h, later reused as y
    bf16_t* q_b      = (bf16_t*)carve((size_t)BATCH * MEMD * 2);
    float*  scores   = (float*) carve((size_t)BATCH * MEMN * 4);   // logits, later reused as sim
    bf16_t* att_b    = (bf16_t*)carve((size_t)BATCH * MEMN * 2);
    bf16_t* hall     = (bf16_t*)carve((size_t)BATCH * HALLN * 2);

    const float inv_sqrt_d = 1.0f / sqrtf((float)MEMD);

    // ---- 1. scaler + features ----
    k_zero2<<<1, 32, 0, stream>>>(gsum);
    k_scaler_pass1<<<BATCH / 256, 256, 0, stream>>>(past, obs, rowsum, rowobs, gsum);
    k_build_feat<<<(BATCH * KPAD) / 256, 256, 0, stream>>>(past, rowsum, rowobs, gsum,
                                                           feat, out_loc, out_scale);

    // ---- 2. weight prep (bf16 convert / transpose / normalize) ----
    k_wb_pad2<<<(KPAD * MEMD) / 256, 256, 0, stream>>>(Wb, Wb_b);
    k_f32_to_bf16<<<(MEMD * MEMD) / 256, 256, 0, stream>>>(Wq, Wq_b, (long)MEMD * MEMD);
    k_transpose_bf16<<<(MEMN * MEMD) / 256, 256, 0, stream>>>(Mem, MemT_b, MEMN, MEMD);
    k_f32_to_bf16<<<(MEMN * MEMD) / 256, 256, 0, stream>>>(Mem, Mem_b, (long)MEMN * MEMD);
    k_ep_rownorm<<<MEMN, 256, 0, stream>>>(Ep, epnorm);
    k_ep_transpose_norm<<<(MEMD * MEMN) / 256, 256, 0, stream>>>(Ep, epnorm, EpT_b);
    k_transpose_bf16<<<((long)MEMD * HALLN) / 256, 256, 0, stream>>>(Wend, WendT_b, MEMD, HALLN);

    const dim3 blk(256);
    const dim3 gN24(BATCH / 128, MEMD / 64);   // N = 1536
    const dim3 gN8 (BATCH / 128, MEMN / 64);   // N = 512

    // ---- 3. h = feat @ Wb + bb (f32) ----
    gemm_bf16_wmma<0><<<gN24, blk, 0, stream>>>(feat, KPAD, Wb_b, MEMD,
                                                hbuf, MEMD, bb, 1.0f, KPAD);
    // ---- 4. pack h -> h_all[:,1] + row norms ----
    k_post_h<<<BATCH, 256, 0, stream>>>(hbuf, hall, rownorm);
    // ---- 5. q = h @ Wq (bf16 out) ----
    gemm_bf16_wmma<1><<<gN24, blk, 0, stream>>>(hall + MEMD, HALLN, Wq_b, MEMD,
                                                q_b, MEMD, nullptr, 1.0f, MEMD);
    // ---- 6. logits = q @ Memory^T / sqrt(d) ----
    gemm_bf16_wmma<0><<<gN8, blk, 0, stream>>>(q_b, MEMD, MemT_b, MEMN,
                                               scores, MEMN, nullptr, inv_sqrt_d, MEMD);
    // ---- 7. att = softmax(logits) ----
    k_softmax512<<<BATCH, 256, 0, stream>>>(scores, att_b);
    // ---- 8. h_memory = att @ Memory  -> h_all[:,0] (bf16, ldc=4608) ----
    gemm_bf16_wmma<1><<<gN24, blk, 0, stream>>>(att_b, MEMN, Mem_b, MEMD,
                                                hall, HALLN, nullptr, 1.0f, MEMN);
    // ---- 9. sim = h @ mn^T (reuse scores buffer) ----
    gemm_bf16_wmma<0><<<gN8, blk, 0, stream>>>(hall + MEMD, HALLN, EpT_b, MEMN,
                                               scores, MEMN, nullptr, 1.0f, MEMD);
    // ---- 10. episodic top-8 retrieval -> h_all[:,2] ----
    k_topk_ep<<<BATCH, 256, 0, stream>>>(scores, rownorm, Ep, hall);
    // ---- 11. y = h_all @ W_end^T + b_end (f32, reuse hbuf) ----
    gemm_bf16_wmma<0><<<gN24, blk, 0, stream>>>(hall, HALLN, WendT_b, MEMD,
                                                hbuf, MEMD, bend, 1.0f, HALLN);
    // ---- 12. StudentT head ----
    k_proj<<<(BATCH * PREDL) / 256, 256, 0, stream>>>(hbuf, Wproj, bproj, out);
}
